// DeepSDSphere_27556510171588
// MI455X (gfx1250) — matrix operand order repaired and checked by reading, not compiled, not run on hardware
//
#include <hip/hip_runtime.h>

typedef _Float16 h8   __attribute__((ext_vector_type(8)));
typedef _Float16 v16h __attribute__((ext_vector_type(16)));
typedef float    v8f  __attribute__((ext_vector_type(8)));

union Frag16 { v16h v; h8 h[2]; };

// ---------------------------------------------------------------------------
// Bilinear 2x upsample (half-pixel centers, clamped, matching jnp reference)
// of a 1-channel plane + concat elevation as channel 1.  dst is NCHW f32.
// ---------------------------------------------------------------------------
__global__ void k_upsample_concat(const float* __restrict__ src,
                                  const float* __restrict__ elev,
                                  float* __restrict__ dst,
                                  int B, int Hin, int Win) {
  const int Hout = Hin * 2, Wout = Win * 2;
  const int total = B * Hout * Wout;
  const int t = blockIdx.x * blockDim.x + threadIdx.x;
  if (t >= total) return;
  const int x = t % Wout;
  const int y = (t / Wout) % Hout;
  const int b = t / (Wout * Hout);

  float sy = (y + 0.5f) * 0.5f - 0.5f;
  sy = fminf(fmaxf(sy, 0.0f), (float)(Hin - 1));
  int r0 = (int)floorf(sy);
  int r1 = r0 + 1; if (r1 > Hin - 1) r1 = Hin - 1;
  const float ty = sy - (float)r0;

  float sx = (x + 0.5f) * 0.5f - 0.5f;
  sx = fminf(fmaxf(sx, 0.0f), (float)(Win - 1));
  int c0 = (int)floorf(sx);
  int c1 = c0 + 1; if (c1 > Win - 1) c1 = Win - 1;
  const float tx = sx - (float)c0;

  const float* sp = src + (size_t)b * Hin * Win;
  const float v00 = sp[r0 * Win + c0];
  const float v01 = sp[r0 * Win + c1];
  const float v10 = sp[r1 * Win + c0];
  const float v11 = sp[r1 * Win + c1];
  const float up = (v00 * (1.f - tx) + v01 * tx) * (1.f - ty)
                 + (v10 * (1.f - tx) + v11 * tx) * ty;

  const size_t hw = (size_t)Hout * Wout;
  float* dp = dst + (size_t)b * 2 * hw;
  const size_t pix = (size_t)y * Wout + x;
  dp[pix]      = up;                                // channel 0: upsampled
  dp[hw + pix] = elev[(size_t)b * hw + pix];        // channel 1: elevation
}

// ---------------------------------------------------------------------------
// Zero the 2-pixel halo of the padded NHWC activation buffer (B,H+4,W+4,64).
// Interior is fully overwritten by k_disco_relu; only the border needs zeros.
// ---------------------------------------------------------------------------
__global__ void k_zero_border(_Float16* __restrict__ a, int B, int H, int W) {
  const int HP = H + 4, WP = W + 4;
  const int total = B * HP * WP;
  const int t = blockIdx.x * blockDim.x + threadIdx.x;
  if (t >= total) return;
  const int x = t % WP;
  const int y = (t / WP) % HP;
  const bool interior = (y >= 2) && (y < H + 2) && (x >= 2) && (x < W + 2);
  if (!interior) {
    const h8 z = {};
    _Float16* p = a + (size_t)t * 64;
#pragma unroll
    for (int i = 0; i < 64; i += 8) *(h8*)(p + i) = z;
  }
}

// ---------------------------------------------------------------------------
// DISCO sparse conv + bias + ReLU.  h:(B,2,P) f32, psi_idx:(P,8), psi_val:
// (9,P,8), w:(64,2,9).  Output written into the INTERIOR of the padded NHWC
// f16 buffer (B,H+4,W+4,64) that feeds the WMMA conv.
// ---------------------------------------------------------------------------
__global__ void k_disco_relu(const float* __restrict__ h,
                             const int* __restrict__ psi_idx,
                             const float* __restrict__ psi_val,
                             const float* __restrict__ w,
                             const float* __restrict__ bias,
                             _Float16* __restrict__ aout,
                             int B, int H, int W) {
  __shared__ float lw[64 * 18];
  __shared__ float lb[64];
  for (int i = threadIdx.x; i < 64 * 18; i += blockDim.x) lw[i] = w[i];
  for (int i = threadIdx.x; i < 64; i += blockDim.x) lb[i] = bias[i];
  __syncthreads();

  const int P = H * W;
  const int t = blockIdx.x * blockDim.x + threadIdx.x;
  const int total = B * P;
  if (t >= total) return;
  const int b = t / P;
  const int p = t - b * P;

  const int4* ip = (const int4*)(psi_idx + (size_t)p * 8);
  const int4 ia = ip[0], ib = ip[1];
  const int idx[8] = { ia.x, ia.y, ia.z, ia.w, ib.x, ib.y, ib.z, ib.w };

  const float* h0 = h + (size_t)b * 2 * P;
  const float* h1 = h0 + P;
  float g0[8], g1[8];
#pragma unroll
  for (int n = 0; n < 8; ++n) { g0[n] = h0[idx[n]]; g1[n] = h1[idx[n]]; }

  float z0[9], z1[9];
#pragma unroll
  for (int k = 0; k < 9; ++k) {
    const float* pv = psi_val + ((size_t)k * P + p) * 8;
    const float4 va = *(const float4*)pv;
    const float4 vb = *(const float4*)(pv + 4);
    z0[k] = g0[0]*va.x + g0[1]*va.y + g0[2]*va.z + g0[3]*va.w
          + g0[4]*vb.x + g0[5]*vb.y + g0[6]*vb.z + g0[7]*vb.w;
    z1[k] = g1[0]*va.x + g1[1]*va.y + g1[2]*va.z + g1[3]*va.w
          + g1[4]*vb.x + g1[5]*vb.y + g1[6]*vb.z + g1[7]*vb.w;
  }

  const int yq = p / W;
  const int xq = p - yq * W;
  _Float16* op = aout + ((size_t)(b * (H + 4) + yq + 2) * (W + 4) + xq + 2) * 64;
  for (int og = 0; og < 64; og += 8) {
    h8 outv;
#pragma unroll
    for (int oo = 0; oo < 8; ++oo) {
      const int o = og + oo;
      const float* wr = &lw[o * 18];
      float acc = lb[o];
#pragma unroll
      for (int k = 0; k < 9; ++k) acc += z0[k] * wr[k];
#pragma unroll
      for (int k = 0; k < 9; ++k) acc += z1[k] * wr[9 + k];
      outv[oo] = (_Float16)fmaxf(acc, 0.0f);
    }
    *(h8*)(op + og) = outv;
  }
}

// ---------------------------------------------------------------------------
// Pre-pack conv weights (32,64,5,5) f32 into the exact B-fragment register
// image for v_wmma_f32_16x16x32_f16: [tap(25)][kb(2)][ntile(2)][lane(32)][16]
// Lane L<16 holds K=0..7,16..23 of column N=L; lane L>=16 holds K=8..15,24..31 (ISA 7.12.2).
// ---------------------------------------------------------------------------
__global__ void k_pack_w(const float* __restrict__ cw, _Float16* __restrict__ wp) {
  const int t = blockIdx.x * blockDim.x + threadIdx.x;
  if (t >= 25 * 2 * 2 * 32 * 16) return;
  const int j    = t & 15;
  const int lane = (t >> 4) & 31;
  const int nt   = (t >> 9) & 1;
  const int kb   = (t >> 10) & 1;
  const int tap  = t >> 11;
  const int kk = ((lane & 16) ? 8 : 0) + (j & 7) + ((j & 8) ? 16 : 0);
  const int c  = kb * 32 + kk;                 // input channel
  const int o  = nt * 16 + (lane & 15);        // output channel
  wp[t] = (_Float16)cw[((size_t)o * 64 + c) * 25 + tap];
}

// ---------------------------------------------------------------------------
// 5x5 SAME conv 64->32 + bias + ReLU as implicit GEMM on WMMA.
// act: zero-padded NHWC f16 (B,H+4,W+4,64); out: NHWC f16 (B,H,W,32).
// Wave tile = 16 pixels x 32 out-channels.  All loops fully unrolled and all
// loads unconditional (halo handles SAME padding; edge lanes clamp their load
// column -- their results are never stored), so the scheduler can clause the
// global loads and pipeline them across the 100 WMMAs.
// ---------------------------------------------------------------------------
__global__ void k_conv5x5_wmma(const _Float16* __restrict__ act,
                               const _Float16* __restrict__ wpack,
                               const float* __restrict__ bias,
                               _Float16* __restrict__ out,
                               int H, int W) {
  const int HP = H + 4, WP = W + 4;
  const int lane = threadIdx.x & 31;
  const int wave = threadIdx.x >> 5;
  const int x0 = blockIdx.x * 128 + wave * 16;
  const int y  = blockIdx.y;
  const int b  = blockIdx.z;
  const int m  = lane & 15;
  const bool hi = (lane & 16) != 0;
  const int x  = x0 + m;
  const int xl = (x < W) ? x : (W - 1);      // clamped load column (store-guarded)
  const int cofs = (hi ? 8 : 0);

  v8f acc0 = {};
  v8f acc1 = {};

#pragma unroll
  for (int ky = 0; ky < 5; ++ky) {
    // padded row index for output row y, tap row ky: (y + ky - 2) + 2 = y + ky
    const _Float16* rowp = act + ((size_t)(b * HP + y + ky) * WP + xl) * 64;
#pragma unroll
    for (int kx = 0; kx < 5; ++kx) {
      const _Float16* ap = rowp + kx * 64;   // padded col index: x + kx
      const int tap = ky * 5 + kx;
#pragma unroll
      for (int kb = 0; kb < 2; ++kb) {
        Frag16 a;
        a.h[0] = *(const h8*)(ap + kb * 32 + cofs);
        a.h[1] = *(const h8*)(ap + kb * 32 + cofs + 16);
        const _Float16* wpb = wpack + (((size_t)(tap * 2 + kb) * 2) * 32 + lane) * 16;
        Frag16 b0, b1;
        b0.h[0] = *(const h8*)(wpb);
        b0.h[1] = *(const h8*)(wpb + 8);
        b1.h[0] = *(const h8*)(wpb + 512);
        b1.h[1] = *(const h8*)(wpb + 520);
        acc0 = __builtin_amdgcn_wmma_f32_16x16x32_f16(false, a.v, false, b0.v,
                                                      (short)0, acc0, false, false);
        acc1 = __builtin_amdgcn_wmma_f32_16x16x32_f16(false, a.v, false, b1.v,
                                                      (short)0, acc1, false, false);
      }
    }
  }

  // C/D layout: lane<16 -> N=lane, rows M=0..7 in regs; lane>=16 -> M=8..15.
  const int n = lane & 15;
  const float bz0 = bias[n];
  const float bz1 = bias[16 + n];
#pragma unroll
  for (int r = 0; r < 8; ++r) {
    const int mm = r + (hi ? 8 : 0);
    const int xs = x0 + mm;
    if (xs < W) {
      _Float16* op = out + (((size_t)(b * H + y) * W + xs) << 5);
      op[n]      = (_Float16)fmaxf(acc0[r] + bz0, 0.0f);
      op[16 + n] = (_Float16)fmaxf(acc1[r] + bz1, 0.0f);
    }
  }
}

// ---------------------------------------------------------------------------
// 5x5 SAME conv 32->1 + bias (no ReLU).  r: NHWC f16; out: (B,1,H,W) f32.
// N=1 makes WMMA 1/16 utilized, and this layer is ~1.5% of FLOPs -> VALU.
// ---------------------------------------------------------------------------
__global__ void k_conv3(const _Float16* __restrict__ r,
                        const float* __restrict__ w,     // (32,5,5): w[c*25+tap]
                        const float* __restrict__ bias,  // (1)
                        float* __restrict__ out,
                        int B, int H, int W) {
  __shared__ float lw[800];
  for (int i = threadIdx.x; i < 800; i += blockDim.x) lw[i] = w[i];
  __syncthreads();
  const int t = blockIdx.x * blockDim.x + threadIdx.x;
  const int total = B * H * W;
  if (t >= total) return;
  const int x = t % W;
  const int y = (t / W) % H;
  const int b = t / (W * H);
  float acc = bias[0];
  for (int ky = 0; ky < 5; ++ky) {
    const int yy = y + ky - 2;
    if (yy < 0 || yy >= H) continue;
    for (int kx = 0; kx < 5; ++kx) {
      const int xx = x + kx - 2;
      if (xx < 0 || xx >= W) continue;
      const int tap = ky * 5 + kx;
      const _Float16* ap = r + (((size_t)(b * H + yy) * W + xx) << 5);
#pragma unroll
      for (int c = 0; c < 32; ++c) acc += (float)ap[c] * lw[c * 25 + tap];
    }
  }
  out[t] = acc;   // NCHW with C=1: (b*H+y)*W+x == t
}

// ---------------------------------------------------------------------------
extern "C" void kernel_launch(void* const* d_in, const int* in_sizes, int n_in,
                              void* d_out, int out_size, void* d_ws, size_t ws_size,
                              hipStream_t stream) {
  const float* x     = (const float*)d_in[0];
  const float* elev0 = (const float*)d_in[1];
  const float* elev1 = (const float*)d_in[2];
  const int*   pidx1 = (const int*)d_in[3];
  const float* pval1 = (const float*)d_in[4];
  const float* w1    = (const float*)d_in[5];
  const float* b1    = (const float*)d_in[6];
  const float* cw2_1 = (const float*)d_in[7];
  const float* cb2_1 = (const float*)d_in[8];
  const float* cw3_1 = (const float*)d_in[9];
  const float* cb3_1 = (const float*)d_in[10];
  const int*   pidx2 = (const int*)d_in[11];
  const float* pval2 = (const float*)d_in[12];
  const float* w2    = (const float*)d_in[13];
  const float* b2    = (const float*)d_in[14];
  const float* cw2_2 = (const float*)d_in[15];
  const float* cb2_2 = (const float*)d_in[16];
  const float* cw3_2 = (const float*)d_in[17];
  const float* cb3_2 = (const float*)d_in[18];

  const int B = 4;
  const int H1 = 180, W1 = 360, P1 = H1 * W1;       // 64800
  const int H2 = 360, W2 = 720, P2 = H2 * W2;       // 259200

  // Workspace layout (all 256B-aligned).  a/r arenas are shared across the
  // two stages (stage-1 tensors are dead before stage-2 writes them).
  char* ws = (char*)d_ws;
  const size_t o_s1  = 0;                                   // (4,1,180,360) f32
  const size_t o_h1  = o_s1  + (size_t)B * P1 * 4;          // (4,2,180,360) f32
  const size_t o_h2  = o_h1  + (size_t)B * 2 * P1 * 4;      // (4,2,360,720) f32
  const size_t o_wp1 = o_h2  + (size_t)B * 2 * P2 * 4;      // packed weights 1
  const size_t o_wp2 = o_wp1 + 102400;                      // packed weights 2
  const size_t o_a   = o_wp2 + 102400;                      // padded NHWC 64ch f16
  const size_t a_big = (size_t)B * (H2 + 4) * (W2 + 4) * 64 * 2;   // 134,930,432
  const size_t o_r   = o_a + a_big;                         // NHWC 32ch f16

  float*     s1   = (float*)(ws + o_s1);
  float*     h1   = (float*)(ws + o_h1);
  float*     h2   = (float*)(ws + o_h2);
  _Float16*  wp1  = (_Float16*)(ws + o_wp1);
  _Float16*  wp2  = (_Float16*)(ws + o_wp2);
  _Float16*  abuf = (_Float16*)(ws + o_a);
  _Float16*  rbuf = (_Float16*)(ws + o_r);

  // Weight packing (independent of activations; issue early).
  k_pack_w<<<200, 256, 0, stream>>>(cw2_1, wp1);
  k_pack_w<<<200, 256, 0, stream>>>(cw2_2, wp2);

  // -------- stage 1 (90x180 -> 180x360) --------
  {
    const int tot  = B * P1;
    const int totp = B * (H1 + 4) * (W1 + 4);
    k_zero_border<<<(totp + 255) / 256, 256, 0, stream>>>(abuf, B, H1, W1);
    k_upsample_concat<<<(tot + 255) / 256, 256, 0, stream>>>(x, elev0, h1, B, 90, 180);
    k_disco_relu<<<(tot + 255) / 256, 256, 0, stream>>>(h1, pidx1, pval1, w1, b1, abuf, B, H1, W1);
    k_conv5x5_wmma<<<dim3(3, H1, B), 256, 0, stream>>>(abuf, wp1, cb2_1, rbuf, H1, W1);
    k_conv3<<<(tot + 255) / 256, 256, 0, stream>>>(rbuf, cw3_1, cb3_1, s1, B, H1, W1);
  }

  // -------- stage 2 (180x360 -> 360x720) --------
  {
    const int tot  = B * P2;
    const int totp = B * (H2 + 4) * (W2 + 4);
    k_zero_border<<<(totp + 255) / 256, 256, 0, stream>>>(abuf, B, H2, W2);
    k_upsample_concat<<<(tot + 255) / 256, 256, 0, stream>>>(s1, elev1, h2, B, H1, W1);
    k_disco_relu<<<(tot + 255) / 256, 256, 0, stream>>>(h2, pidx2, pval2, w2, b2, abuf, B, H2, W2);
    k_conv5x5_wmma<<<dim3(6, H2, B), 256, 0, stream>>>(abuf, wp2, cb2_2, rbuf, H2, W2);
    k_conv3<<<(tot + 255) / 256, 256, 0, stream>>>(rbuf, cw3_2, cb3_2, (float*)d_out, B, H2, W2);
  }

  (void)in_sizes; (void)n_in; (void)out_size; (void)ws_size;
}
